// TTConvEinsum_66546223284785
// MI455X (gfx1250) — matrix operand order repaired and checked by reading, not compiled, not run on hardware
//
#include <hip/hip_runtime.h>
#include <hip/hip_bf16.h>

typedef __attribute__((ext_vector_type(16))) __bf16 v16bf;
typedef __attribute__((ext_vector_type(8)))  float  v8f;
typedef unsigned short ushort_t;

__device__ __forceinline__ ushort_t f2bf(float f) {
    unsigned int u = __builtin_bit_cast(unsigned int, f);
    unsigned int r = u + 0x7FFFu + ((u >> 16) & 1u);   // round-to-nearest-even
    return (ushort_t)(r >> 16);
}

// ---------------------------------------------------------------------------
// Core-merge kernels (tiny flops, f32): core2 x core3 -> W23 ; x core1 -> W123 ;
// x core0 -> transposed full conv weight WfullT[o][k' = q*256 + c] in bf16.
// ---------------------------------------------------------------------------
__global__ __launch_bounds__(256) void tt_merge23(
    const float* __restrict__ core2, const float* __restrict__ core3,
    float* __restrict__ W23) {
    int idx = blockIdx.x * 256 + threadIdx.x;          // 32768 = 8*4*8*8*16
    int r1 = idx & 15;
    int j2 = (idx >> 4) & 7;
    int j1 = (idx >> 7) & 7;
    int i2 = (idx >> 10) & 3;
    int i1 = idx >> 12;
    const float* c2 = core2 + (((i1 * 8 + j1) * 16 + r1) << 4); // [i1][j1][r1][r2]
    const float* c3 = core3 + ((i2 * 8 + j2) << 4);             // [i2][j2][r2]
    float s = 0.f;
#pragma unroll
    for (int r2 = 0; r2 < 16; ++r2) s += c2[r2] * c3[r2];
    W23[idx] = s;   // layout [i1][i2][j1][j2][r1] = i1*4096+i2*1024+j1*128+j2*16+r1
}

__global__ __launch_bounds__(256) void tt_merge123(
    const float* __restrict__ core1, const float* __restrict__ W23,
    float* __restrict__ W123) {
    int idx = blockIdx.x * 256 + threadIdx.x;          // 2097152 = 256*512*16
    int r0 = idx & 15;
    int o  = (idx >> 4) & 511;
    int c  = idx >> 13;
    int j2 = o & 7, j1 = (o >> 3) & 7, j0 = o >> 6;
    int i2 = c & 3, i1 = (c >> 2) & 7, i0 = c >> 5;
    const float* c1 = core1 + i0 * 2048 + j0 * 256 + r0 * 16;           // [i0][j0][r0][r1]
    const float* w  = W23   + i1 * 4096 + i2 * 1024 + j1 * 128 + j2 * 16;
    float s = 0.f;
#pragma unroll
    for (int r1 = 0; r1 < 16; ++r1) s += c1[r1] * w[r1];
    W123[idx] = s;  // layout [c][o][r0] = c*8192 + o*16 + r0  (== idx)
}

__global__ __launch_bounds__(256) void tt_mergefull(
    const float* __restrict__ core0, const float* __restrict__ W123,
    ushort_t* __restrict__ WfullT) {
    const int c = threadIdx.x;   // 0..255
    const int o = blockIdx.x;    // 0..511
    const int q = blockIdx.y;    // 0..8  (kh*3+kw)
    const float* w = W123 + c * 8192 + o * 16;
    float s = 0.f;
#pragma unroll
    for (int r0 = 0; r0 < 16; ++r0) s += core0[r0 * 9 + q] * w[r0];
    WfullT[(size_t)o * 2304 + q * 256 + c] = f2bf(s);  // tap-major K, write coalesced
}

// ---------------------------------------------------------------------------
// Implicit-im2col GEMM: out[p, o] = sum_k' A[p,k'] * WfullT[o,k']
//   M = 16384 pixels (b,h,w), N = 512, K = 2304 = 9 taps x 256 channels.
//   K order k' = q*256 + c: every 32-chunk has ONE tap -> no per-element
//   index math, immediate-offset global loads, vector LDS staging.
// Block: 256 thr (8 waves), tile 128(M) x 128(N), bf16 WMMA, f32 acc.
// ---------------------------------------------------------------------------
#define ASTR 40   // LDS row stride in halfs (80B: 16B aligned, bank-spread)

__global__ __launch_bounds__(256) void tt_conv_gemm(
    const float* __restrict__ x, const ushort_t* __restrict__ wfullT,
    float* __restrict__ out) {
    __shared__ ushort_t Ash[128 * ASTR];   // [row(pixel) 0..127][kk 0..31]
    __shared__ ushort_t Bsh[128 * ASTR];   // [col(outch) 0..127][kk 0..31]

    const int tid  = threadIdx.x;
    const int n0   = blockIdx.x * 128;     // 4 N-blocks
    const int p0   = blockIdx.y * 128;     // 128 M-blocks (never crosses batch)
    const int lane = tid & 31;
    const int wv   = tid >> 5;

    // A staging: thread -> fixed row, 16 consecutive kk
    const int arowS  = tid & 127;          // pixel row in tile
    const int ahalfS = tid >> 7;           // kk base = ahalfS*16
    const int p  = p0 + arowS;
    const int bb = p >> 12;
    const int hh = (p >> 6) & 63;
    const int ww = p & 63;

    // B staging: thread -> col = tid>>1, 16 consecutive kk (seg = tid&1)
    const int bcolS = tid >> 1;
    const int bsegS = tid & 1;

    // WMMA fragment addressing (documented 16-bit A 16x32 / B 32x16 layouts)
    const int arow  = wv * 16 + (lane & 15);
    const int ahalf = (lane < 16) ? 0 : 8;
    const int bhalf = (lane < 16) ? 0 : 8;

    const v8f zero = {0.f, 0.f, 0.f, 0.f, 0.f, 0.f, 0.f, 0.f};
    v8f acc[8];
#pragma unroll
    for (int n = 0; n < 8; ++n) acc[n] = zero;

    for (int q = 0; q < 9; ++q) {          // conv tap (uniform per chunk)
        const int kh  = q / 3;
        const int kw  = q - 3 * kh;
        const int hin = hh + kh - 1;
        const int win = ww + kw - 1;
        const bool valid = ((unsigned)hin < 64u) && ((unsigned)win < 64u);
        // base for this thread's row/tap; element c adds c*4096 floats
        const float* xq = x + (((size_t)bb) << 20) + ((size_t)(ahalfS * 16) << 12)
                            + (hin << 6) + win;

        for (int cb = 0; cb < 8; ++cb) {
            const int k0 = q * 256 + cb * 32;
            __syncthreads();
            // --- stage A: 16 loads at immediate offsets j*16KB, pack to bf16 ---
            float tv[16];
            if (valid) {
                const float* xp = xq + ((size_t)(cb * 32) << 12);
#pragma unroll
                for (int j = 0; j < 16; ++j) tv[j] = xp[(size_t)j << 12];
            } else {
#pragma unroll
                for (int j = 0; j < 16; ++j) tv[j] = 0.f;
            }
            unsigned int pk[8];
#pragma unroll
            for (int j = 0; j < 8; ++j)
                pk[j] = (unsigned)f2bf(tv[2 * j]) |
                        ((unsigned)f2bf(tv[2 * j + 1]) << 16);
            uint4* adst = (uint4*)&Ash[arowS * ASTR + ahalfS * 16];
            uint4 av0; av0.x = pk[0]; av0.y = pk[1]; av0.z = pk[2]; av0.w = pk[3];
            uint4 av1; av1.x = pk[4]; av1.y = pk[5]; av1.z = pk[6]; av1.w = pk[7];
            adst[0] = av0;
            adst[1] = av1;
            // --- stage B: 2x16B contiguous loads from transposed weight ---
            const ushort_t* wp = wfullT + (size_t)(n0 + bcolS) * 2304 + k0 + bsegS * 16;
            uint4 bv0 = *(const uint4*)wp;
            uint4 bv1 = *(const uint4*)(wp + 8);
            uint4* bdst = (uint4*)&Bsh[bcolS * ASTR + bsegS * 16];
            bdst[0] = bv0;
            bdst[1] = bv1;
            __syncthreads();
            // --- fragments + 8 WMMAs ---
            union { uint4 u[2]; v16bf v; } af;
            af.u[0] = *(const uint4*)&Ash[arow * ASTR + ahalf];
            af.u[1] = *(const uint4*)&Ash[arow * ASTR + 16 + ahalf];
#pragma unroll
            for (int n = 0; n < 8; ++n) {
                union { uint4 u[2]; v16bf v; } bf;
                const int bcol = n * 16 + (lane & 15);
                bf.u[0] = *(const uint4*)&Bsh[bcol * ASTR + bhalf];
                bf.u[1] = *(const uint4*)&Bsh[bcol * ASTR + bhalf + 16];
                acc[n] = __builtin_amdgcn_wmma_f32_16x16x32_bf16(
                    false, af.v, false, bf.v, (short)0, acc[n], false, false);
            }
        }
    }

    // --- epilogue: C layout lane(0-15)=N, vgpr r = M r / r+8 -> NCHW stores ---
    const int rlo = wv * 16 + ((lane < 16) ? 0 : 8);
#pragma unroll
    for (int n = 0; n < 8; ++n) {
        int o = n0 + n * 16 + (lane & 15);
#pragma unroll
        for (int r = 0; r < 8; ++r) {
            int pp = p0 + rlo + r;
            int b  = pp >> 12;
            int hw = pp & 4095;
            out[(((b << 9) + o) << 12) + hw] = acc[n][r];
        }
    }
}

extern "C" void kernel_launch(void* const* d_in, const int* in_sizes, int n_in,
                              void* d_out, int out_size, void* d_ws, size_t ws_size,
                              hipStream_t stream) {
    const float* x     = (const float*)d_in[0];   // (4,256,64,64)
    const float* core0 = (const float*)d_in[1];   // (16,1,3,3)
    const float* core1 = (const float*)d_in[2];   // (8,8,16,16)
    const float* core2 = (const float*)d_in[3];   // (8,8,16,16)
    const float* core3 = (const float*)d_in[4];   // (4,8,16,1)
    float* out = (float*)d_out;                   // (4,512,64,64)

    // workspace layout (~10.9 MB)
    float*    W23    = (float*)d_ws;                                   // 32768 f32
    float*    W123   = (float*)((char*)d_ws + 131072);                 // 2097152 f32
    ushort_t* WfullT = (ushort_t*)((char*)d_ws + 131072 + 8388608);    // 512*2304 bf16

    tt_merge23  <<<dim3(128),    256, 0, stream>>>(core2, core3, W23);
    tt_merge123 <<<dim3(8192),   256, 0, stream>>>(core1, W23, W123);
    tt_mergefull<<<dim3(512, 9), 256, 0, stream>>>(core0, W123, WfullT);
    tt_conv_gemm<<<dim3(4, 128), 256, 0, stream>>>(x, WfullT, out);
}
